// PtrNet_37684043055877
// MI455X (gfx1250) — matrix operand order repaired and checked by reading, not compiled, not run on hardware
//
#include <hip/hip_runtime.h>
#include <hip/hip_bf16.h>

// ---------------- problem constants ----------------
#define BB   512
#define NN   100
#define DD   256
#define FD   1024   // 4*D

typedef __attribute__((ext_vector_type(16))) _Float16 v16h;
typedef __attribute__((ext_vector_type(8)))  float    v8f;

union Frag16 { v16h v; uint4 u[2]; };

__device__ __forceinline__ float sigf(float x) { return 1.0f / (1.0f + __expf(-x)); }

// =====================================================================
// WMMA f16 GEMM:  C[M][N] = concat_K(A0,A1)[M][K] * Wt[N][K]^T + bias[N]
// A0/A1 are f32 (lda0/lda1 row strides), converted to f16 in LDS via VALU.
// Wt is pre-transposed/converted f16, row-major [N][K]; its tile is staged
// with CDNA5 async copies (global_load_async_to_lds_b128, ASYNCcnt).
// Block tile 64(M) x 128(N), 256 threads = 8 waves, wave tile 32x32
// (2x2 of v_wmma_f32_16x16x32_f16). K, K0 must be multiples of 32.
// =====================================================================
__global__ __launch_bounds__(256)
void gemm_wmma_f16(const float* __restrict__ A0, int lda0,
                   const float* __restrict__ A1, int lda1, int K0, int K,
                   const _Float16* __restrict__ Wt,
                   const float* __restrict__ bias,
                   float* __restrict__ C, int ldc)
{
    __shared__ _Float16 As[64][32];
    __shared__ _Float16 Bs[128][32];

    const int tid  = threadIdx.x;
    const int bm   = blockIdx.x * 64;
    const int bn   = blockIdx.y * 128;
    const int lane = tid & 31;
    const int wid  = tid >> 5;
    const int wm   = (wid >> 2) * 32;   // 0 or 32
    const int wn   = (wid & 3) * 32;    // 0,32,64,96

    v8f acc[2][2] = {};

    const int arow = tid >> 2;          // 0..63
    const int acg  = (tid & 3) * 8;     // 0,8,16,24
    const int brow = tid >> 1;          // 0..127
    const int bcg  = (tid & 1) * 16;    // 0,16

    const int akb = (lane < 16) ? 0 : 8;
    const int bkb = (lane < 16) ? 0 : 8;
    const int mr  = lane & 15;

    // wave-relative LDS byte offset of this thread's B-tile slot
    // (generic pointer low 32 bits == LDS offset per aperture mapping)
    const unsigned bs_lds = (unsigned)(unsigned long long)&Bs[brow][bcg];

    for (int k0 = 0; k0 < K; k0 += 32) {
        __syncthreads();
        // ---- async-stage B tile: memory -> LDS directly, no VGPRs ----
        // INST_OFFSET applies to both LDS and global addresses, so one base
        // plus offset:0 / offset:16 moves this thread's 32-byte row segment.
        {
            const _Float16* gsrc = Wt + (size_t)(bn + brow) * K + k0 + bcg;
            asm volatile("global_load_async_to_lds_b128 %0, %1, off"
                         :: "v"(bs_lds), "v"(gsrc) : "memory");
            asm volatile("global_load_async_to_lds_b128 %0, %1, off offset:16"
                         :: "v"(bs_lds), "v"(gsrc) : "memory");
        }
        // ---- stage A tile (f32 -> f16 convert through VGPRs) ----
        {
            const int kg = k0 + acg;
            const float* src = (kg < K0)
                ? (A0 + (size_t)(bm + arow) * lda0 + kg)
                : (A1 + (size_t)(bm + arow) * lda1 + (kg - K0));
            __builtin_prefetch(src + 32, 0, 0);   // speculative next-chunk prefetch
            float4 f0 = *(const float4*)(src);
            float4 f1 = *(const float4*)(src + 4);
            union { _Float16 h[8]; uint4 u; } t;
            t.h[0] = (_Float16)f0.x; t.h[1] = (_Float16)f0.y;
            t.h[2] = (_Float16)f0.z; t.h[3] = (_Float16)f0.w;
            t.h[4] = (_Float16)f1.x; t.h[5] = (_Float16)f1.y;
            t.h[6] = (_Float16)f1.z; t.h[7] = (_Float16)f1.w;
            *(uint4*)&As[arow][acg] = t.u;
        }
        // our own async B copies must land before the barrier releases readers
        asm volatile("s_wait_asynccnt 0x0" ::: "memory");
        __syncthreads();

        Frag16 af[2], bf[2];
#pragma unroll
        for (int ti = 0; ti < 2; ++ti) {
            const int row = wm + 16 * ti + mr;
            af[ti].u[0] = *(const uint4*)&As[row][akb];
            af[ti].u[1] = *(const uint4*)&As[row][akb + 16];
        }
#pragma unroll
        for (int tj = 0; tj < 2; ++tj) {
            const int col = wn + 16 * tj + mr;
            bf[tj].u[0] = *(const uint4*)&Bs[col][bkb];
            bf[tj].u[1] = *(const uint4*)&Bs[col][bkb + 16];
        }
#pragma unroll
        for (int ti = 0; ti < 2; ++ti)
#pragma unroll
            for (int tj = 0; tj < 2; ++tj)
                acc[ti][tj] = __builtin_amdgcn_wmma_f32_16x16x32_f16(
                    false, af[ti].v, false, bf[tj].v,
                    (short)0, acc[ti][tj], false, false);
    }

    // epilogue: C/D layout: VGPR r -> M = r + 8*(lane>=16), N = lane&15
    const int rofs = (lane < 16) ? 0 : 8;
    const int cn   = lane & 15;
#pragma unroll
    for (int ti = 0; ti < 2; ++ti)
#pragma unroll
        for (int tj = 0; tj < 2; ++tj) {
            const int col = bn + wn + 16 * tj + cn;
            const float bv = bias ? bias[col] : 0.0f;
#pragma unroll
            for (int r = 0; r < 8; ++r) {
                const int row = bm + wm + 16 * ti + rofs + r;
                C[(size_t)row * ldc + col] = acc[ti][tj][r] + bv;
            }
        }
}

// =====================================================================
// Weight prep: transpose + f32->f16
// =====================================================================
__global__ void prep_lstm_w(const float* __restrict__ eWih, const float* __restrict__ eWhh,
                            const float* __restrict__ dWih, const float* __restrict__ dWhh,
                            _Float16* __restrict__ We, _Float16* __restrict__ Wd)
{
    const int total = FD * 512;
    int idx = blockIdx.x * blockDim.x + threadIdx.x;
    if (idx >= 2 * total) return;
    const int sel = idx / total;
    const int rem = idx - sel * total;
    const int n = rem >> 9;     // /512  (output col 0..1023)
    const int k = rem & 511;    // concat-K (x:0..255, h:256..511)
    const float* Wih = sel ? dWih : eWih;
    const float* Whh = sel ? dWhh : eWhh;
    const float v = (k < DD) ? Wih[k * FD + n] : Whh[(k - DD) * FD + n];
    (sel ? Wd : We)[n * 512 + k] = (_Float16)v;
}

__global__ void prep_proj_w(const float* __restrict__ glWq, const float* __restrict__ glWr,
                            const float* __restrict__ ptWq, const float* __restrict__ ptWr,
                            _Float16* __restrict__ TglWq, _Float16* __restrict__ TglWr,
                            _Float16* __restrict__ TptWq, _Float16* __restrict__ TptWr)
{
    int idx = blockIdx.x * blockDim.x + threadIdx.x;
    if (idx >= 4 * DD * DD) return;
    const int sel = idx >> 16;
    const int rem = idx & 65535;
    const int n = rem >> 8;
    const int k = rem & 255;
    const float* src = (sel == 0) ? glWq : (sel == 1) ? glWr : (sel == 2) ? ptWq : ptWr;
    _Float16*   dst = (sel == 0) ? TglWq : (sel == 1) ? TglWr : (sel == 2) ? TptWq : TptWr;
    dst[n * DD + k] = (_Float16)src[k * DD + n];
}

__global__ void prep_bias(const float* __restrict__ ebih, const float* __restrict__ ebhh,
                          const float* __restrict__ dbih, const float* __restrict__ dbhh,
                          float* __restrict__ be, float* __restrict__ bd)
{
    int i = blockIdx.x * blockDim.x + threadIdx.x;
    if (i < FD)            be[i]      = ebih[i] + ebhh[i];
    else if (i < 2 * FD)   bd[i - FD] = dbih[i - FD] + dbhh[i - FD];
}

// =====================================================================
// Embedding: embd[b,n,d] = sum_f problems[b,n,f] * W_embed[f,d]
// =====================================================================
__global__ void embed_k(const float* __restrict__ problems,
                        const float* __restrict__ We,
                        float* __restrict__ embd)
{
    int idx = blockIdx.x * blockDim.x + threadIdx.x;
    if (idx >= BB * NN * DD) return;
    const int bn = idx >> 8;
    const int d  = idx & 255;
    const float* p = problems + (size_t)bn * 4;
    embd[idx] = p[0] * We[d] + p[1] * We[DD + d] + p[2] * We[2 * DD + d] + p[3] * We[3 * DD + d];
}

__global__ void zero_hc(float* __restrict__ h, float* __restrict__ c)
{
    int idx = blockIdx.x * blockDim.x + threadIdx.x;
    if (idx < BB * DD) { h[idx] = 0.0f; c[idx] = 0.0f; }
}

__global__ void init_dec(float* __restrict__ x, const float* __restrict__ d0,
                         int* __restrict__ mask)
{
    int idx = blockIdx.x * blockDim.x + threadIdx.x;
    if (idx < BB * DD) x[idx] = d0[idx & 255];
    if (idx < BB * NN) mask[idx] = 0;
}

// =====================================================================
// LSTM gate pointwise (torch gate order i,f,g,o)
// =====================================================================
__global__ void lstm_pointwise(const float* __restrict__ gates,
                               float* __restrict__ h, float* __restrict__ c,
                               float* __restrict__ enc_out, int enc_stride)
{
    int idx = blockIdx.x * blockDim.x + threadIdx.x;
    if (idx >= BB * DD) return;
    const int b = idx >> 8;
    const int d = idx & 255;
    const float* g = gates + (size_t)b * FD;
    const float gi = g[d], gf = g[DD + d], gg = g[2 * DD + d], go = g[3 * DD + d];
    const float c2 = sigf(gf) * c[idx] + sigf(gi) * tanhf(gg);
    const float h2 = sigf(go) * tanhf(c2);
    c[idx] = c2;
    h[idx] = h2;
    if (enc_out) enc_out[(size_t)b * enc_stride + d] = h2;
}

// =====================================================================
// u[b,n] = 10 * sum_d V[d]*tanh(qp[b,d] + ref[b,n,d]) ; one wave per (b,n)
// =====================================================================
__global__ __launch_bounds__(256)
void attn_u(const float* __restrict__ qp, const float* __restrict__ ref,
            const float* __restrict__ V, float* __restrict__ u)
{
    const int pair = blockIdx.x * 8 + (threadIdx.x >> 5);
    const int lane = threadIdx.x & 31;
    const int b = pair / NN;
    const float* q = qp + (size_t)b * DD;
    const float* r = ref + (size_t)pair * DD;
    float acc = 0.0f;
#pragma unroll
    for (int j = 0; j < 8; ++j) {
        const int d = lane + j * 32;
        acc += V[d] * tanhf(q[d] + r[d]);
    }
#pragma unroll
    for (int m = 16; m; m >>= 1) acc += __shfl_xor(acc, m, 32);
    if (lane == 0) u[pair] = 10.0f * acc;
}

// =====================================================================
// Glimpse: masked softmax over n, then q2[b,:] = sum_n p[n]*ref[b,n,:]
// =====================================================================
__global__ __launch_bounds__(256)
void glimpse_k(const float* __restrict__ u, const int* __restrict__ mask,
               const float* __restrict__ ref, float* __restrict__ q2)
{
    __shared__ float p[NN];
    __shared__ float red[2];
    const int b = blockIdx.x, tid = threadIdx.x;
    if (tid < NN) p[tid] = mask[b * NN + tid] ? -3.4e38f : u[b * NN + tid];
    __syncthreads();
    if (tid == 0) {
        float mx = -3.4e38f;
        for (int n = 0; n < NN; ++n) mx = fmaxf(mx, p[n]);
        red[0] = mx;
    }
    __syncthreads();
    if (tid < NN) p[tid] = __expf(p[tid] - red[0]);
    __syncthreads();
    if (tid == 0) {
        float s = 0.0f;
        for (int n = 0; n < NN; ++n) s += p[n];
        red[1] = 1.0f / s;
    }
    __syncthreads();
    const float inv = red[1];
    float acc = 0.0f;
    const float* rb = ref + (size_t)b * NN * DD + tid;
    for (int n = 0; n < NN; ++n) acc += p[n] * rb[n * DD];
    q2[b * DD + tid] = acc * inv;
}

// =====================================================================
// Pointer head: masked softmax, greedy argmax, record prob/action,
// update mask, gather next decoder input from embd.
// =====================================================================
__global__ __launch_bounds__(256)
void ptr_final(const float* __restrict__ u, int* __restrict__ mask,
               const float* __restrict__ embd, float* __restrict__ x,
               float* __restrict__ probs_out, float* __restrict__ act_out, int t)
{
    __shared__ float lv[NN];
    __shared__ int sa;
    const int b = blockIdx.x, tid = threadIdx.x;
    if (tid < NN) lv[tid] = mask[b * NN + tid] ? -3.4e38f : u[b * NN + tid];
    __syncthreads();
    if (tid == 0) {
        float best = -3.4e38f; int a = 0;
        for (int n = 0; n < NN; ++n) if (lv[n] > best) { best = lv[n]; a = n; }
        float s = 0.0f;
        for (int n = 0; n < NN; ++n) s += __expf(lv[n] - best);
        sa = a;
        probs_out[b * NN + t] = 1.0f / s;       // softmax value at argmax
        act_out[b * NN + t]   = (float)a;
        mask[b * NN + a] = 1;
    }
    __syncthreads();
    const int a = sa;
    x[b * DD + tid] = embd[((size_t)b * NN + a) * DD + tid];
}

// =====================================================================
// Host-side orchestration
// =====================================================================
extern "C" void kernel_launch(void* const* d_in, const int* in_sizes, int n_in,
                              void* d_out, int out_size, void* d_ws, size_t ws_size,
                              hipStream_t stream)
{
    const float* problems = (const float*)d_in[0];
    const float* W_embed  = (const float*)d_in[1];
    const float* enc_Wih  = (const float*)d_in[2];
    const float* enc_Whh  = (const float*)d_in[3];
    const float* enc_bih  = (const float*)d_in[4];
    const float* enc_bhh  = (const float*)d_in[5];
    const float* dec_Wih  = (const float*)d_in[6];
    const float* dec_Whh  = (const float*)d_in[7];
    const float* dec_bih  = (const float*)d_in[8];
    const float* dec_bhh  = (const float*)d_in[9];
    const float* gl_Wq    = (const float*)d_in[10];
    const float* gl_bq    = (const float*)d_in[11];
    const float* gl_Wr    = (const float*)d_in[12];
    const float* gl_br    = (const float*)d_in[13];
    const float* gl_V     = (const float*)d_in[14];
    const float* pt_Wq    = (const float*)d_in[15];
    const float* pt_bq    = (const float*)d_in[16];
    const float* pt_Wr    = (const float*)d_in[17];
    const float* pt_br    = (const float*)d_in[18];
    const float* pt_V     = (const float*)d_in[19];
    const float* dec_in0  = (const float*)d_in[20];

    float* probs_out = (float*)d_out;                   // [B,N]
    float* act_out   = (float*)d_out + (size_t)BB * NN; // [B,N] (actions as float)

    // workspace carve-out
    char* base = (char*)d_ws;
    size_t off = 0;
    auto alloc = [&](size_t bytes) -> void* {
        void* p = base + off;
        off += (bytes + 255) & ~(size_t)255;
        return p;
    };
    float*    embd  = (float*)alloc((size_t)BB * NN * DD * 4);
    float*    encs  = (float*)alloc((size_t)BB * NN * DD * 4);
    float*    glref = (float*)alloc((size_t)BB * NN * DD * 4);
    float*    ptref = (float*)alloc((size_t)BB * NN * DD * 4);
    float*    gates = (float*)alloc((size_t)BB * FD * 4);
    float*    h     = (float*)alloc((size_t)BB * DD * 4);
    float*    c     = (float*)alloc((size_t)BB * DD * 4);
    float*    x     = (float*)alloc((size_t)BB * DD * 4);
    float*    qp    = (float*)alloc((size_t)BB * DD * 4);
    float*    q2    = (float*)alloc((size_t)BB * DD * 4);
    float*    qp2   = (float*)alloc((size_t)BB * DD * 4);
    float*    u     = (float*)alloc((size_t)BB * NN * 4);
    float*    u2    = (float*)alloc((size_t)BB * NN * 4);
    int*      mask  = (int*)alloc((size_t)BB * NN * 4);
    _Float16* We    = (_Float16*)alloc((size_t)FD * 512 * 2);
    _Float16* Wd    = (_Float16*)alloc((size_t)FD * 512 * 2);
    _Float16* TglWq = (_Float16*)alloc((size_t)DD * DD * 2);
    _Float16* TglWr = (_Float16*)alloc((size_t)DD * DD * 2);
    _Float16* TptWq = (_Float16*)alloc((size_t)DD * DD * 2);
    _Float16* TptWr = (_Float16*)alloc((size_t)DD * DD * 2);
    float*    be    = (float*)alloc(FD * 4);
    float*    bd    = (float*)alloc(FD * 4);

    // ---- prep ----
    prep_lstm_w<<<(2 * FD * 512 + 255) / 256, 256, 0, stream>>>(enc_Wih, enc_Whh, dec_Wih, dec_Whh, We, Wd);
    prep_proj_w<<<(4 * DD * DD + 255) / 256, 256, 0, stream>>>(gl_Wq, gl_Wr, pt_Wq, pt_Wr, TglWq, TglWr, TptWq, TptWr);
    prep_bias<<<(2 * FD + 255) / 256, 256, 0, stream>>>(enc_bih, enc_bhh, dec_bih, dec_bhh, be, bd);
    embed_k<<<(BB * NN * DD + 255) / 256, 256, 0, stream>>>(problems, W_embed, embd);
    zero_hc<<<(BB * DD + 255) / 256, 256, 0, stream>>>(h, c);

    // ---- encoder LSTM over node axis ----
    for (int t = 0; t < NN; ++t) {
        gemm_wmma_f16<<<dim3(BB / 64, FD / 128), 256, 0, stream>>>(
            embd + (size_t)t * DD, NN * DD,   // x_t rows strided by N*D
            h, DD, DD, 2 * DD, We, be, gates, FD);
        lstm_pointwise<<<(BB * DD + 255) / 256, 256, 0, stream>>>(
            gates, h, c, encs + (size_t)t * DD, NN * DD);
    }

    // ---- ref projections (step-invariant) ----
    gemm_wmma_f16<<<dim3(BB * NN / 64, DD / 128), 256, 0, stream>>>(
        encs, DD, encs, DD, DD, DD, TglWr, gl_br, glref, DD);
    gemm_wmma_f16<<<dim3(BB * NN / 64, DD / 128), 256, 0, stream>>>(
        encs, DD, encs, DD, DD, DD, TptWr, pt_br, ptref, DD);

    init_dec<<<(BB * DD + 255) / 256, 256, 0, stream>>>(x, dec_in0, mask);

    // ---- decoder loop (h,c carry over from encoder final state) ----
    for (int t = 0; t < NN; ++t) {
        gemm_wmma_f16<<<dim3(BB / 64, FD / 128), 256, 0, stream>>>(
            x, DD, h, DD, DD, 2 * DD, Wd, bd, gates, FD);
        lstm_pointwise<<<(BB * DD + 255) / 256, 256, 0, stream>>>(
            gates, h, c, nullptr, 0);

        // glimpse attention
        gemm_wmma_f16<<<dim3(BB / 64, DD / 128), 256, 0, stream>>>(
            h, DD, h, DD, DD, DD, TglWq, gl_bq, qp, DD);
        attn_u<<<BB * NN / 8, 256, 0, stream>>>(qp, glref, gl_V, u);
        glimpse_k<<<BB, 256, 0, stream>>>(u, mask, glref, q2);

        // pointer attention
        gemm_wmma_f16<<<dim3(BB / 64, DD / 128), 256, 0, stream>>>(
            q2, DD, q2, DD, DD, DD, TptWq, pt_bq, qp2, DD);
        attn_u<<<BB * NN / 8, 256, 0, stream>>>(qp2, ptref, pt_V, u2);
        ptr_final<<<BB, 256, 0, stream>>>(u2, mask, embd, x, probs_out, act_out, t);
    }
}
